// Model_32091995635825
// MI455X (gfx1250) — matrix operand order repaired and checked by reading, not compiled, not run on hardware
//
#include <hip/hip_runtime.h>

typedef __attribute__((ext_vector_type(16))) _Float16 v16h;
typedef __attribute__((ext_vector_type(8)))  float    v8f;

// ---------------------------------------------------------------------------
// Utility: grid-stride zero fill (graph-capture safe, no memset needed)
// ---------------------------------------------------------------------------
__global__ void zero_kernel(float* __restrict__ p, size_t n) {
  size_t t = (size_t)blockIdx.x * blockDim.x + threadIdx.x;
  size_t stride = (size_t)gridDim.x * blockDim.x;
  for (; t < n; t += stride) p[t] = 0.0f;
}

// ---------------------------------------------------------------------------
// Degree accumulation: deg[dst] += edge_weight
// ---------------------------------------------------------------------------
__global__ void deg_kernel(const float* __restrict__ w, const int* __restrict__ dst,
                           float* __restrict__ deg, int E) {
  int e = blockIdx.x * blockDim.x + threadIdx.x;
  if (e >= E) return;
  atomicAdd(&deg[dst[e]], w[e]);
}

// dinv = rsqrt(deg + 1), self_coef = dinv^2
__global__ void nodecoef_kernel(const float* __restrict__ deg, float* __restrict__ dinv,
                                float* __restrict__ selfc, int N) {
  int i = blockIdx.x * blockDim.x + threadIdx.x;
  if (i >= N) return;
  float r = rsqrtf(deg[i] + 1.0f);
  dinv[i] = r;
  selfc[i] = r * r;
}

// coef[e] = dinv[src]*w*dinv[dst]
__global__ void edgecoef_kernel(const float* __restrict__ dinv, const float* __restrict__ w,
                                const int* __restrict__ src, const int* __restrict__ dst,
                                float* __restrict__ coef, int E) {
  int e = blockIdx.x * blockDim.x + threadIdx.x;
  if (e >= E) return;
  coef[e] = dinv[src[e]] * w[e] * dinv[dst[e]];
}

// ---------------------------------------------------------------------------
// Build h0: [x(8) | atom_emb[id](32) | zeros(24)] -> f16 [N][64]
// ---------------------------------------------------------------------------
__global__ void embed_kernel(const float* __restrict__ x, const float* __restrict__ atom_emb,
                             _Float16* __restrict__ h, int N) {
  int i = blockIdx.x * blockDim.x + threadIdx.x;
  if (i >= N) return;
  const float* xi = x + (size_t)i * 8;
  _Float16* hi = h + (size_t)i * 64;
  int id = (int)xi[0];
  id = id < 0 ? 0 : (id > 19 ? 19 : id);
  #pragma unroll
  for (int f = 0; f < 8; ++f) hi[f] = (_Float16)xi[f];
  const float* em = atom_emb + id * 32;
  #pragma unroll 8
  for (int j = 0; j < 32; ++j) hi[8 + j] = (_Float16)em[j];
  #pragma unroll 8
  for (int j = 40; j < 64; ++j) hi[j] = (_Float16)0.0f;
}

// ---------------------------------------------------------------------------
// Weight convert: W [kdim][64] f32 -> Wt [64][64] f16 transposed, K zero-padded
// ---------------------------------------------------------------------------
__global__ void wconv_kernel(const float* __restrict__ W, _Float16* __restrict__ Wt, int kdim) {
  int t = blockIdx.x * blockDim.x + threadIdx.x;
  if (t >= 64 * 64) return;
  int n = t >> 6, k = t & 63;
  Wt[n * 64 + k] = (k < kdim) ? (_Float16)W[k * 64 + n] : (_Float16)0.0f;
}

// ---------------------------------------------------------------------------
// WMMA GEMM: out[N][64] f32 = h[N][64] f16 @ W (Wt = W^T f16 [64][64])
// Block = 128 threads = 4 waves; wave w owns column strip n0 = w*16 and
// processes MTILES=4 row tiles (block covers 64 rows), reusing the B
// fragments (weights) across all row tiles.
//
// A-fragment (16x32 f16): lanes 0-15 rows M, dwords v0..3 = K {0..7} pairs,
//   v4..7 = K {16..23}; lanes 16-31 same rows, K +8 / +24. -> two uint4 loads.
// B-fragment (32x16 f16): lane%16 = col, same K order as A (lanes 0-15 K
//   0..7,16..23; lanes 16-31 K 8..15,24..31) -> two uint4 loads from transposed weights.
// rows must be a multiple of 16 (true here: N = 200000); tiles are guarded
// at tile granularity only, so stores are branch-free immediate-offset b32s.
// ---------------------------------------------------------------------------
#define MTILES 4
__global__ __launch_bounds__(128) void gemm_wmma_kernel(
    const _Float16* __restrict__ h, const _Float16* __restrict__ wt,
    float* __restrict__ out, int rows) {
  const int wave = threadIdx.x >> 5;
  const int lane = threadIdx.x & 31;
  const int m0 = blockIdx.x * (16 * MTILES);
  if (m0 >= rows) return;
  const int n0 = wave * 16;
  const bool laneLo = lane < 16;
  const int l15 = lane & 15;
  const int colB = n0 + l15;
  const int rbase = laneLo ? 0 : 8;
  const int kAoff = laneLo ? 0 : 8;   // A K-offset within 32-chunk
  const int kBoff = laneLo ? 0 : 8;   // B K-offset within 32-chunk

  // B fragments for both K-chunks, loaded once per wave
  union { uint4 u[2]; v16h v; } ub0, ub1;
  const _Float16* wcol = wt + (size_t)colB * 64;
  ub0.u[0] = *(const uint4*)(wcol + kBoff);
  ub0.u[1] = *(const uint4*)(wcol + kBoff + 16);
  ub1.u[0] = *(const uint4*)(wcol + 32 + kBoff);
  ub1.u[1] = *(const uint4*)(wcol + 32 + kBoff + 16);

  #pragma unroll
  for (int t = 0; t < MTILES; ++t) {
    const int mt = m0 + t * 16;
    if (mt >= rows) break;           // rows % 16 == 0, so no partial tiles
    const int rowA = mt + l15;
    const _Float16* hrow = h + (size_t)rowA * 64;
    union { uint4 u[2]; v16h v; } ua0, ua1;
    ua0.u[0] = *(const uint4*)(hrow + kAoff);
    ua0.u[1] = *(const uint4*)(hrow + kAoff + 16);
    ua1.u[0] = *(const uint4*)(hrow + 32 + kAoff);
    ua1.u[1] = *(const uint4*)(hrow + 32 + kAoff + 16);

    v8f acc = {};
    acc = __builtin_amdgcn_wmma_f32_16x16x32_f16(
        false, ua0.v, false, ub0.v, (short)0, acc, false, false);
    acc = __builtin_amdgcn_wmma_f32_16x16x32_f16(
        false, ua1.v, false, ub1.v, (short)0, acc, false, false);

    // D layout: VGPR r -> row mt + rbase + r, col = colB. Branch-free stores.
    float* po = out + (size_t)(mt + rbase) * 64 + colB;
    #pragma unroll
    for (int r = 0; r < 8; ++r) po[r * 64] = acc[r];
  }
}

// ---------------------------------------------------------------------------
// Edge scatter: agg[dst] += coef[e] * hw[src]; 16 threads per edge, float4 each
// ---------------------------------------------------------------------------
__global__ void scatter_kernel(const float* __restrict__ hw, const float* __restrict__ coef,
                               const int* __restrict__ src, const int* __restrict__ dst,
                               float* __restrict__ agg, int E) {
  int t = blockIdx.x * blockDim.x + threadIdx.x;
  int e = t >> 4;
  if (e >= E) return;
  int q = (t & 15) << 2;
  float c = coef[e];
  int s = src[e], d = dst[e];
  float4 v = *(const float4*)(hw + (size_t)s * 64 + q);
  float* a = agg + (size_t)d * 64 + q;
  atomicAdd(a + 0, v.x * c);
  atomicAdd(a + 1, v.y * c);
  atomicAdd(a + 2, v.z * c);
  atomicAdd(a + 3, v.w * c);
}

// ---------------------------------------------------------------------------
// Finalize conv layer: h_next = relu(agg + self_coef*hw + b)  (f16 out)
// ---------------------------------------------------------------------------
__global__ void finalize_kernel(const float* __restrict__ agg, const float* __restrict__ hw,
                                const float* __restrict__ selfc, const float* __restrict__ b,
                                _Float16* __restrict__ hout, int n64) {
  int t = blockIdx.x * blockDim.x + threadIdx.x;
  if (t >= n64) return;
  int i = t >> 6, f = t & 63;
  float v = agg[t] + selfc[i] * hw[t] + b[f];
  hout[t] = (_Float16)(v > 0.0f ? v : 0.0f);
}

// ---------------------------------------------------------------------------
// Pooling: pooled[batch[i]] += hw[i]; cnt[batch[i]] += 1 (bias folded in head)
// ---------------------------------------------------------------------------
__global__ void pool_kernel(const float* __restrict__ hw, const int* __restrict__ batch,
                            float* __restrict__ pooled, float* __restrict__ cnt, int n64) {
  int t = blockIdx.x * blockDim.x + threadIdx.x;
  if (t >= n64) return;
  int i = t >> 6, f = t & 63;
  int g = batch[i];
  atomicAdd(&pooled[(size_t)g * 64 + f], hw[t]);
  if (f == 0) atomicAdd(&cnt[g], 1.0f);
}

// ---------------------------------------------------------------------------
// MLP head: one block (128 thr) per graph.
// z0 = [pooled/cnt + bg (64) | relu(prot_emb) (10)] -> 128 -> 96 -> 32 -> sigmoid
// ---------------------------------------------------------------------------
__global__ __launch_bounds__(128) void head_kernel(
    const float* __restrict__ pooled, const float* __restrict__ cnt,
    const float* __restrict__ bg, const int* __restrict__ protein,
    const float* __restrict__ prot_emb,
    const float* __restrict__ L1w, const float* __restrict__ L1b,
    const float* __restrict__ L2w, const float* __restrict__ L2b,
    const float* __restrict__ L3w, const float* __restrict__ L3b,
    const float* __restrict__ L4w, const float* __restrict__ L4b,
    float* __restrict__ out) {
  __shared__ float z0[80];
  __shared__ float z1[128];
  __shared__ float z2[96];
  __shared__ float z3[32];
  const int g = blockIdx.x;
  const int t = threadIdx.x;
  float inv = 1.0f / cnt[g];
  if (t < 64) {
    z0[t] = pooled[(size_t)g * 64 + t] * inv + bg[t];
  } else if (t < 74) {
    float v = prot_emb[protein[g] * 10 + (t - 64)];
    z0[t] = v > 0.0f ? v : 0.0f;
  }
  __syncthreads();
  {
    float s = L1b[t];
    #pragma unroll 2
    for (int i = 0; i < 74; ++i) s += z0[i] * L1w[i * 128 + t];
    z1[t] = s > 0.0f ? s : 0.0f;
  }
  __syncthreads();
  if (t < 96) {
    float s = L2b[t];
    #pragma unroll 4
    for (int i = 0; i < 128; ++i) s += z1[i] * L2w[i * 96 + t];
    z2[t] = s > 0.0f ? s : 0.0f;
  }
  __syncthreads();
  if (t < 32) {
    float s = L3b[t];
    #pragma unroll 4
    for (int i = 0; i < 96; ++i) s += z2[i] * L3w[i * 32 + t];
    z3[t] = s > 0.0f ? s : 0.0f;
  }
  __syncthreads();
  if (t == 0) {
    float s = L4b[0];
    #pragma unroll
    for (int i = 0; i < 32; ++i) s += z3[i] * L4w[i];
    out[g] = 1.0f / (1.0f + expf(-s));
  }
}

// ---------------------------------------------------------------------------
extern "C" void kernel_launch(void* const* d_in, const int* in_sizes, int n_in,
                              void* d_out, int out_size, void* d_ws, size_t ws_size,
                              hipStream_t stream) {
  const float* x        = (const float*)d_in[0];
  const int*   ei       = (const int*)  d_in[1];
  const float* ew       = (const float*)d_in[2];
  const int*   batch    = (const int*)  d_in[3];
  const int*   protein  = (const int*)  d_in[4];
  const float* atom_emb = (const float*)d_in[5];
  const float* prot_emb = (const float*)d_in[6];
  const float* Ws[5] = {(const float*)d_in[7],  (const float*)d_in[9],
                        (const float*)d_in[11], (const float*)d_in[13],
                        (const float*)d_in[15]};
  const float* bs[5] = {(const float*)d_in[8],  (const float*)d_in[10],
                        (const float*)d_in[12], (const float*)d_in[14],
                        (const float*)d_in[16]};
  const float* L1w = (const float*)d_in[17]; const float* L1b = (const float*)d_in[18];
  const float* L2w = (const float*)d_in[19]; const float* L2b = (const float*)d_in[20];
  const float* L3w = (const float*)d_in[21]; const float* L3b = (const float*)d_in[22];
  const float* L4w = (const float*)d_in[23]; const float* L4b = (const float*)d_in[24];

  const int N = in_sizes[0] / 8;
  const int E = in_sizes[2];
  const int G = in_sizes[4];
  const int* src = ei;
  const int* dst = ei + E;

  // ---- workspace carve (256B-aligned) ----
  size_t off = 0;
  auto carve = [&](size_t bytes) -> void* {
    void* p = (char*)d_ws + off;
    off += (bytes + 255) & ~(size_t)255;
    return p;
  };
  float*    deg    = (float*)   carve((size_t)N * 4);
  float*    dinv   = (float*)   carve((size_t)N * 4);
  float*    selfc  = (float*)   carve((size_t)N * 4);
  float*    coef   = (float*)   carve((size_t)E * 4);
  _Float16* h16    = (_Float16*)carve((size_t)N * 64 * 2);
  float*    hw     = (float*)   carve((size_t)N * 64 * 4);
  float*    agg    = (float*)   carve((size_t)N * 64 * 4);
  _Float16* Wt     = (_Float16*)carve((size_t)5 * 64 * 64 * 2);
  float*    pooled = (float*)   carve((size_t)G * 64 * 4);
  float*    cnt    = (float*)   carve((size_t)G * 4);
  (void)ws_size; (void)n_in; (void)out_size;

  const dim3 b256(256);
  const size_t n64 = (size_t)N * 64;

  // normalization coefficients
  zero_kernel<<<1024, b256, 0, stream>>>(deg, (size_t)N);
  deg_kernel<<<(E + 255) / 256, b256, 0, stream>>>(ew, dst, deg, E);
  nodecoef_kernel<<<(N + 255) / 256, b256, 0, stream>>>(deg, dinv, selfc, N);
  edgecoef_kernel<<<(E + 255) / 256, b256, 0, stream>>>(dinv, ew, src, dst, coef, E);

  // input features + weight conversion
  embed_kernel<<<(N + 255) / 256, b256, 0, stream>>>(x, atom_emb, h16, N);
  const int kdim[5] = {40, 64, 64, 64, 64};
  for (int l = 0; l < 5; ++l)
    wconv_kernel<<<16, b256, 0, stream>>>(Ws[l], Wt + (size_t)l * 64 * 64, kdim[l]);

  // 4 GCN conv layers
  const int gemmBlocks = (N + 16 * MTILES - 1) / (16 * MTILES);
  for (int l = 0; l < 4; ++l) {
    gemm_wmma_kernel<<<gemmBlocks, 128, 0, stream>>>(h16, Wt + (size_t)l * 64 * 64, hw, N);
    zero_kernel<<<2048, b256, 0, stream>>>(agg, n64);
    scatter_kernel<<<(E * 16 + 255) / 256, b256, 0, stream>>>(hw, coef, src, dst, agg, E);
    finalize_kernel<<<(int)((n64 + 255) / 256), b256, 0, stream>>>(agg, hw, selfc, bs[l],
                                                                   h16, (int)n64);
  }

  // output projection (Wg, no conv) -> pooling (bg folded into head)
  gemm_wmma_kernel<<<gemmBlocks, 128, 0, stream>>>(h16, Wt + (size_t)4 * 64 * 64, hw, N);
  zero_kernel<<<256, b256, 0, stream>>>(pooled, (size_t)G * 64);
  zero_kernel<<<16, b256, 0, stream>>>(cnt, (size_t)G);
  pool_kernel<<<(int)((n64 + 255) / 256), b256, 0, stream>>>(hw, batch, pooled, cnt, (int)n64);

  // MLP head
  head_kernel<<<G, 128, 0, stream>>>(pooled, cnt, bs[4], protein, prot_emb,
                                     L1w, L1b, L2w, L2b, L3w, L3b, L4w, L4b,
                                     (float*)d_out);
}
